// E3Conv_76639396430365
// MI455X (gfx1250) — matrix operand order repaired and checked by reading, not compiled, not run on hardware
//
#include <hip/hip_runtime.h>
#include <hip/hip_bf16.h>

// CDNA5 / gfx1250, wave32. All matmul-shaped work goes through
// v_wmma_f32_16x16x32_f16.

typedef __attribute__((ext_vector_type(16))) _Float16 v16h;
typedef __attribute__((ext_vector_type(8)))  _Float16 v8h;
typedef __attribute__((ext_vector_type(8)))  float    v8f;

#define WAVES_PER_BLK 4
#define BLK (WAVES_PER_BLK * 32)

// packed-f16 weight regions inside d_ws (offsets in halves)
// layout per region: [kchunk][ntile][lane(32)][q(16)]
#define W1_OFF 0        // fcw1: K=32(pad of 10), N=64  -> 1*4*32*16 = 2048
#define W2_OFF 2048     // fcw2: K=64, N=64             -> 2*4*32*16 = 4096
#define W3_OFF 6144     // fcw3: K=64, N=64             -> 4096
#define W4_OFF 10240    // fused fcw4 ⊗ TP: K=4096, N=32-> 128*2*32*16 = 131072
#define WTOT   141312

__device__ __forceinline__ float silu_act(float x) {
    return x / (1.0f + __expf(-x));
}

// ---------------------------------------------------------------------------
// Pre-pack all radial weights into WMMA B-fragment order (f16), folding in
// every scalar norm:  l1: 10^-1/2, l2/l3: 64^-1/2, l4: 64^-1/2 * (8*8)^-1/2.
// B layout (32x16 K x N per 16x16x32 WMMA): lanes 0-15 hold K {0..7,16..23},
// lanes 16-31 hold K {8..15,24..31}; column n = lane&15 (+16*ntile).
// ---------------------------------------------------------------------------
__global__ void __launch_bounds__(256) prep_weights(
    const float* __restrict__ fcw1, const float* __restrict__ fcw2,
    const float* __restrict__ fcw3, const float* __restrict__ fcw4,
    _Float16* __restrict__ wpack)
{
    int idx = blockIdx.x * 256 + threadIdx.x;
    if (idx >= WTOT) return;
    float val = 0.0f;
    if (idx < W2_OFF) {                         // fcw1 (10x64, K padded to 32)
        int local = idx;
        int q = local & 15, lane = (local >> 4) & 31, nt = local >> 9;
        int kglob = ((q & 8) << 1) + ((lane & 16) >> 1) + (q & 7);
        int col = nt * 16 + (lane & 15);
        if (kglob < 10) val = fcw1[kglob * 64 + col] * 0.31622776601683794f;
    } else if (idx < W3_OFF) {                  // fcw2 (64x64)
        int local = idx - W2_OFF;
        int q = local & 15, lane = (local >> 4) & 31;
        int nt = (local >> 9) & 3, kc = local >> 11;
        int kglob = kc * 32 + (((q & 8) << 1) + ((lane & 16) >> 1) + (q & 7));
        int col = nt * 16 + (lane & 15);
        val = fcw2[kglob * 64 + col] * 0.125f;
    } else if (idx < W4_OFF) {                  // fcw3 (64x64)
        int local = idx - W3_OFF;
        int q = local & 15, lane = (local >> 4) & 31;
        int nt = (local >> 9) & 3, kc = local >> 11;
        int kglob = kc * 32 + (((q & 8) << 1) + ((lane & 16) >> 1) + (q & 7));
        int col = nt * 16 + (lane & 15);
        val = fcw3[kglob * 64 + col] * 0.125f;
    } else {                                    // fused fcw4 / tensor-product
        int local = idx - W4_OFF;
        int q = local & 15, lane = (local >> 4) & 31;
        int t = (local >> 9) & 1, c = local >> 10;       // c = 0..127 K-chunks
        int k  = c >> 1;                                 // radial feature 0..63
        int kin = ((q & 8) << 1) + ((lane & 16) >> 1) + (q & 7);
        int uj = (c & 1) * 32 + kin;                     // uj = u*8+j, 0..63
        int u = uj >> 3, j = uj & 7;
        int n = t * 16 + (lane & 15);                    // output column v'
        if (n < 16)      val = fcw4[k * 1792 + u * 128 + j * 16 + n];
        else if (n < 24) val = fcw4[k * 1792 + 1024 + u * 64 + j * 8 + (n - 16)];
        else if (n < 28) val = fcw4[k * 1792 + 1536 + u * 32 + j * 4 + (n - 24)];
        val *= 0.015625f;   // 64^-1/2 (last radial layer) * (8*8)^-1/2 (TP norm)
    }
    wpack[idx] = (_Float16)val;
}

// ---------------------------------------------------------------------------
// Node embedding MLP: Ai[n,0:8] = mlp(emb_table[A[n]])   (tiny: 66 MFLOP)
// ---------------------------------------------------------------------------
__global__ void __launch_bounds__(128) node_mlp_kernel(
    const int* __restrict__ A, const float* __restrict__ emb_table,
    const float* __restrict__ fw1, const float* __restrict__ fb1,
    const float* __restrict__ fw2, const float* __restrict__ fb2,
    const float* __restrict__ fw3, const float* __restrict__ fb3,
    float* __restrict__ Ai, int N)
{
    int n = blockIdx.x * 128 + threadIdx.x;
    if (n >= N) return;
    int a = A[n];
    float x[16];
    #pragma unroll
    for (int i = 0; i < 16; ++i) x[i] = emb_table[a * 16 + i];
    float h1[64];
    for (int j = 0; j < 64; ++j) {
        float s = fb1[j];
        #pragma unroll
        for (int i = 0; i < 16; ++i) s += x[i] * fw1[i * 64 + j];
        h1[j] = silu_act(s);
    }
    float h2[32];
    for (int j = 0; j < 32; ++j) {
        float s = fb2[j];
        for (int k = 0; k < 64; ++k) s += h1[k] * fw2[k * 32 + j];
        h2[j] = silu_act(s);
    }
    for (int j = 0; j < 8; ++j) {
        float s = fb3[j];
        for (int k = 0; k < 32; ++k) s += h2[k] * fw3[k * 8 + j];
        Ai[n * 8 + j] = s;
    }
}

__global__ void __launch_bounds__(256) zero_kernel(float* __restrict__ p, int n)
{
    int i = blockIdx.x * 256 + threadIdx.x;
    if (i < n) p[i] = 0.0f;
}

__global__ void __launch_bounds__(256) finalize_kernel(
    const float* __restrict__ accb, const float* __restrict__ cntb,
    float* __restrict__ out, int N)
{
    int i = blockIdx.x * 256 + threadIdx.x;
    if (i >= N * 60) return;
    int n = i / 60;
    out[i] = accb[i] / fmaxf(cntb[n], 1.0f);
}

// ---------------------------------------------------------------------------
// Main fused edge kernel. 4 waves/block; each wave owns 32 edges.
// Phase 0: geometry, sh, Gaussian basis, ai/aj gather -> LDS (per-wave slice,
//          same-wave LDS ordering makes barriers unnecessary).
// Phase 1: radial MLP layers as 16x16x32 f16 WMMAs (activations ping-pong in
//          LDS as f16, SILU applied on f32 accumulators).
// Phase 2: fused GEMM t = z @ W4 with K=4096; A-fragments built on the fly as
//          h3[k] * (ai⊗aj) (8 v_pk_mul_f16), B streamed pre-packed from L2.
// Epilogue: sh-broadcast + global f32 atomics into segment accumulators.
// ---------------------------------------------------------------------------
__global__ void __launch_bounds__(BLK) edge_kernel(
    const float* __restrict__ pos, const int* __restrict__ batch,
    const int* __restrict__ esrc, const int* __restrict__ edst,
    const float* __restrict__ shifts, const float* __restrict__ cell,
    const float* __restrict__ Ai, const _Float16* __restrict__ wpack,
    float* __restrict__ accb, float* __restrict__ cntb, int E)
{
    __shared__ __align__(16) _Float16 s_act[WAVES_PER_BLK][32][72]; // f16 acts
    __shared__ float s_ai[WAVES_PER_BLK][32][8];
    __shared__ float s_aj[WAVES_PER_BLK][32][8];
    __shared__ float s_sh[WAVES_PER_BLK][32][8];   // sh[1..8]
    __shared__ int   s_dst[WAVES_PER_BLK][32];

    const int tid  = threadIdx.x;
    const int w    = tid >> 5;
    const int lane = tid & 31;
    const int mrow = lane & 15;
    const int hlf  = lane >> 4;          // 0: lanes 0-15, 1: lanes 16-31

    const int e_glob = blockIdx.x * BLK + tid;
    const bool valid = (e_glob < E);

    // ---------------- phase 0: per-edge scalar prep ----------------
    if (valid) {
        int src = esrc[e_glob], dst = edst[e_glob];
        s_dst[w][lane] = dst;
        float sx = shifts[e_glob * 3 + 0];
        float sy = shifts[e_glob * 3 + 1];
        float sz = shifts[e_glob * 3 + 2];
        const float* C = cell + (size_t)batch[src] * 9;
        float vx = pos[dst * 3 + 0] - pos[src * 3 + 0] + sx * C[0] + sy * C[3] + sz * C[6];
        float vy = pos[dst * 3 + 1] - pos[src * 3 + 1] + sx * C[1] + sy * C[4] + sz * C[7];
        float vz = pos[dst * 3 + 2] - pos[src * 3 + 2] + sx * C[2] + sy * C[5] + sz * C[8];
        float l2 = vx * vx + vy * vy + vz * vz + 1e-12f;
        float rinv = rsqrtf(l2);
        float len = l2 * rinv;
        float x = vx * rinv, y = vy * rinv, z = vz * rinv;
        const float s3 = 1.7320508075688772f, s5 = 2.2360679774997896f,
                    s15 = 3.872983346207417f;
        s_sh[w][lane][0] = s3 * x;
        s_sh[w][lane][1] = s3 * y;
        s_sh[w][lane][2] = s3 * z;
        s_sh[w][lane][3] = s15 * x * z;
        s_sh[w][lane][4] = s15 * x * y;
        s_sh[w][lane][5] = s5 * (y * y - 0.5f * (x * x + z * z));
        s_sh[w][lane][6] = s15 * y * z;
        s_sh[w][lane][7] = 0.5f * s15 * (z * z - x * x);
        // Gaussian soft-one-hot (already * sqrt(10)/1.12), padded to K=32
        const float inv_step = 11.0f / 4.0f;
        #pragma unroll
        for (int i = 0; i < 10; ++i) {
            float d = (len - (float)(i + 1) * (4.0f / 11.0f)) * inv_step;
            s_act[w][lane][i] = (_Float16)(__expf(-d * d) * 2.8234622f);
        }
        #pragma unroll
        for (int i = 10; i < 32; ++i) s_act[w][lane][i] = (_Float16)0.0f;
        #pragma unroll
        for (int u = 0; u < 8; ++u) s_ai[w][lane][u] = Ai[(size_t)src * 8 + u];
        #pragma unroll
        for (int j = 0; j < 8; ++j) s_aj[w][lane][j] = Ai[(size_t)dst * 8 + j];
        atomicAdd(&cntb[dst], 1.0f);
    } else {
        s_dst[w][lane] = 0;
        #pragma unroll
        for (int i = 0; i < 32; ++i) s_act[w][lane][i] = (_Float16)0.0f;
        #pragma unroll
        for (int u = 0; u < 8; ++u) {
            s_ai[w][lane][u] = 0.0f; s_aj[w][lane][u] = 0.0f; s_sh[w][lane][u] = 0.0f;
        }
    }
    __syncthreads();

    // A-fragment loader per ISA 7.12.2: lanes 0-15 K {0..7,16..23},
    // lanes 16-31 K {8..15,24..31}; row M = lane&15.
    auto load_afrag = [&](int e, int kc) -> v16h {
        int base = kc * 32 + (hlf ? 8 : 0);
        v8h a0 = *(const v8h*)&s_act[w][e][base];
        v8h a1 = *(const v8h*)&s_act[w][e][base + 16];
        v16h a;
        #pragma unroll
        for (int i = 0; i < 8; ++i) { a[i] = a0[i]; a[8 + i] = a1[i]; }
        return a;
    };

    // One radial layer: act(16..32 x Kc*32) @ W(Kc*32 x 64) -> SILU -> s_act
    auto run_layer = [&](const _Float16* wb, int kchunks) {
        const v16h* B = (const v16h*)wb;
        v8f acc[2][4] = {};
        for (int kc = 0; kc < kchunks; ++kc) {
            v16h bf[4];
            #pragma unroll
            for (int nt = 0; nt < 4; ++nt) bf[nt] = B[(kc * 4 + nt) * 32 + lane];
            #pragma unroll
            for (int s = 0; s < 2; ++s) {
                v16h a = load_afrag(mrow + 16 * s, kc);
                #pragma unroll
                for (int nt = 0; nt < 4; ++nt)
                    acc[s][nt] = __builtin_amdgcn_wmma_f32_16x16x32_f16(
                        false, a, false, bf[nt], (short)0, acc[s][nt], false, false);
            }
        }
        // SILU_NORM * silu(.), store back as f16 (C layout: N=lane&15, M=r+8*hlf)
        #pragma unroll
        for (int s = 0; s < 2; ++s)
            #pragma unroll
            for (int nt = 0; nt < 4; ++nt) {
                int j = nt * 16 + mrow;
                #pragma unroll
                for (int r = 0; r < 8; ++r) {
                    int e = r + hlf * 8 + 16 * s;
                    s_act[w][e][j] = (_Float16)(1.679177f * silu_act(acc[s][nt][r]));
                }
            }
    };

    run_layer(wpack + W1_OFF, 1);   // 10->64
    run_layer(wpack + W2_OFF, 2);   // 64->64
    run_layer(wpack + W3_OFF, 2);   // 64->64   (h3 now in s_act, f16)

    // -------- phase 2: fused K=4096 GEMM (weight-gen + tensor product) ------
    // Per-lane outer-product halves: lanes 0-15 own u in {0,2,4,6},
    // lanes 16-31 own u in {1,3,5,7}; Lo covers uj 0..31, Hi covers 32..63.
    v16h outLo[2], outHi[2];
    #pragma unroll
    for (int s = 0; s < 2; ++s) {
        int e = mrow + 16 * s;
        float ajv[8];
        #pragma unroll
        for (int j = 0; j < 8; ++j) ajv[j] = s_aj[w][e][j];
        float a0 = s_ai[w][e][hlf + 0];
        float a2 = s_ai[w][e][hlf + 2];
        float a4 = s_ai[w][e][hlf + 4];
        float a6 = s_ai[w][e][hlf + 6];
        #pragma unroll
        for (int j = 0; j < 8; ++j) {
            outLo[s][j]     = (_Float16)(a0 * ajv[j]);
            outLo[s][8 + j] = (_Float16)(a2 * ajv[j]);
            outHi[s][j]     = (_Float16)(a4 * ajv[j]);
            outHi[s][8 + j] = (_Float16)(a6 * ajv[j]);
        }
    }

    v8f acc[2][2] = {};
    const v16h* B4 = (const v16h*)(wpack + W4_OFF);
    for (int k = 0; k < 64; ++k) {
        _Float16 hk0 = s_act[w][mrow     ][k];
        _Float16 hk1 = s_act[w][mrow + 16][k];
        #pragma unroll
        for (int sub = 0; sub < 2; ++sub) {
            int c = k * 2 + sub;
            v16h b0 = B4[(c * 2 + 0) * 32 + lane];
            v16h b1 = B4[(c * 2 + 1) * 32 + lane];
            if (c < 124)
                __builtin_prefetch((const void*)&B4[(c * 2 + 8) * 32 + lane], 0, 0);
            v16h a0 = (sub ? outHi[0] : outLo[0]) * hk0;   // 8x v_pk_mul_f16
            v16h a1 = (sub ? outHi[1] : outLo[1]) * hk1;
            acc[0][0] = __builtin_amdgcn_wmma_f32_16x16x32_f16(
                false, a0, false, b0, (short)0, acc[0][0], false, false);
            acc[0][1] = __builtin_amdgcn_wmma_f32_16x16x32_f16(
                false, a0, false, b1, (short)0, acc[0][1], false, false);
            acc[1][0] = __builtin_amdgcn_wmma_f32_16x16x32_f16(
                false, a1, false, b0, (short)0, acc[1][0], false, false);
            acc[1][1] = __builtin_amdgcn_wmma_f32_16x16x32_f16(
                false, a1, false, b1, (short)0, acc[1][1], false, false);
        }
    }

    // -------- epilogue: sh broadcast + segment accumulation (atomics) -------
    const int ebase = blockIdx.x * BLK + w * 32;
    #pragma unroll
    for (int s = 0; s < 2; ++s)
        #pragma unroll
        for (int t = 0; t < 2; ++t) {
            int vp = t * 16 + mrow;          // output channel v' (0..27 valid)
            if (vp < 28) {
                #pragma unroll
                for (int r = 0; r < 8; ++r) {
                    int el = r + hlf * 8 + 16 * s;
                    if (ebase + el < E) {
                        float val = acc[s][t][r];
                        float* dp = accb + (size_t)s_dst[w][el] * 60;
                        if (vp < 16) {
                            atomicAdd(dp + vp, val);                  // o0 (sh0=1)
                        } else if (vp < 24) {
                            int v = vp - 16;
                            #pragma unroll
                            for (int m = 0; m < 3; ++m)               // o1 * sh[1+m]
                                atomicAdd(dp + 16 + v * 3 + m, val * s_sh[w][el][m]);
                        } else {
                            int v = vp - 24;
                            #pragma unroll
                            for (int m = 0; m < 5; ++m)               // o2 * sh[4+m]
                                atomicAdd(dp + 40 + v * 5 + m, val * s_sh[w][el][3 + m]);
                        }
                    }
                }
            }
        }
}

// ---------------------------------------------------------------------------
extern "C" void kernel_launch(void* const* d_in, const int* in_sizes, int n_in,
                              void* d_out, int out_size, void* d_ws, size_t ws_size,
                              hipStream_t stream)
{
    const float* pos       = (const float*)d_in[0];
    const int*   A         = (const int*)  d_in[1];
    const int*   batch     = (const int*)  d_in[2];
    const int*   esrc      = (const int*)  d_in[3];
    const int*   edst      = (const int*)  d_in[4];
    const float* shifts    = (const float*)d_in[5];
    const float* cell      = (const float*)d_in[6];
    const float* emb_table = (const float*)d_in[7];
    const float* fw1 = (const float*)d_in[8];
    const float* fb1 = (const float*)d_in[9];
    const float* fw2 = (const float*)d_in[10];
    const float* fb2 = (const float*)d_in[11];
    const float* fw3 = (const float*)d_in[12];
    const float* fb3 = (const float*)d_in[13];
    const float* fcw1 = (const float*)d_in[14];
    const float* fcw2 = (const float*)d_in[15];
    const float* fcw3 = (const float*)d_in[16];
    const float* fcw4 = (const float*)d_in[17];

    const int N = in_sizes[0] / 3;
    const int E = in_sizes[3];

    char* ws = (char*)d_ws;
    _Float16* wpack = (_Float16*)ws;                        // 282,624 B
    size_t off = ((size_t)WTOT * 2 + 255) & ~(size_t)255;
    float* Ai = (float*)(ws + off);                         // N*8 f32
    off += (size_t)N * 8 * sizeof(float);
    off = (off + 255) & ~(size_t)255;
    float* accb = (float*)(ws + off);                       // N*60 f32
    float* cntb = accb + (size_t)N * 60;                    // N f32 (contiguous)

    prep_weights<<<(WTOT + 255) / 256, 256, 0, stream>>>(fcw1, fcw2, fcw3, fcw4, wpack);
    zero_kernel<<<(N * 61 + 255) / 256, 256, 0, stream>>>(accb, N * 61);
    node_mlp_kernel<<<(N + 127) / 128, 128, 0, stream>>>(
        A, emb_table, fw1, fb1, fw2, fb2, fw3, fb3, Ai, N);
    edge_kernel<<<(E + BLK - 1) / BLK, BLK, 0, stream>>>(
        pos, batch, esrc, edst, shifts, cell, Ai, wpack, accb, cntb, E);
    finalize_kernel<<<(N * 60 + 255) / 256, 256, 0, stream>>>(
        accb, cntb, (float*)d_out, N);
}
